// CompositionalEmbedding_2250562863572
// MI455X (gfx1250) — matrix operand order repaired and checked by reading, not compiled, not run on hardware
//
#include <hip/hip_runtime.h>
#include <hip/hip_bf16.h>

// Problem constants (from the reference)
#define NB    32            // codebooks
#define NC    64            // codewords per codebook
#define KDIM  (NB * NC)     // 2048 = GEMM K
#define DDIM  256           // embedding dim = GEMM N
#define MTILE 32            // tokens per block = GEMM M tile (2 WMMA M-tiles)
#define APAD  8             // LDS row padding (f16) -> row stride 2056 = 1028 dwords == 4 mod 64

typedef __attribute__((ext_vector_type(16))) _Float16 v16h;
typedef __attribute__((ext_vector_type(8)))  _Float16 v8h;
typedef __attribute__((ext_vector_type(8)))  float    v8f;

// ---------------------------------------------------------------------------
// Prologue: convert + transpose codebook once per launch.
//   codebook (2048 x 256 f32, [k][n])  ->  cbT (256 x 2048 f16, [n][k]) in d_ws
// 65536 threads; each handles one n-column and 8 consecutive k's -> one b128 store.
// ---------------------------------------------------------------------------
__global__ __launch_bounds__(256)
void convert_codebook_kernel(const float* __restrict__ cb,
                             _Float16*    __restrict__ cbT)
{
    const int t  = blockIdx.x * 256 + threadIdx.x;   // 0..65535
    const int n  = t & (DDIM - 1);                   // 0..255 (coalesced loads)
    const int k0 = (t >> 8) * 8;                     // 0..2040 step 8
    v8h v;
    #pragma unroll
    for (int i = 0; i < 8; ++i)
        v[i] = (_Float16)cb[(size_t)(k0 + i) * DDIM + n];
    *(v8h*)(cbT + (size_t)n * KDIM + k0) = v;        // contiguous 16B store
}

// ---------------------------------------------------------------------------
// Fused gather -> softmax(axis = codebook) -> (32 x 2048)x(2048 x 256) WMMA GEMM
// ---------------------------------------------------------------------------
__global__ __launch_bounds__(256)
void comp_embed_fused_kernel(const int*      __restrict__ tokens,
                             const float*    __restrict__ code,
                             const _Float16* __restrict__ cbT,   // [n][k] f16
                             float*          __restrict__ out)
{
    // f16 softmax weights for 32 tokens: 32 * 2056 * 2B = 131.6 KB (<= 320 KB/WG)
    __shared__ _Float16 w_lds[MTILE][KDIM + APAD];

    const int t   = threadIdx.x;   // 0..255
    const int blk = blockIdx.x;    // 0..1023

    // ---------------- Phase 1: gather + softmax over the codebook axis ----------
    {
        const int mloc  = t >> 3;      // local token 0..31 (8 threads per token)
        const int lane8 = t & 7;
        const int tok   = tokens[blk * MTILE + mloc];
        const float* __restrict__ row = code + (size_t)tok * KDIM;

        for (int j = 0; j < 8; ++j) {
            const int c = lane8 + 8 * j;          // codeword slot 0..63
            float v[NB];
            float mx = -3.402823466e38f;
            #pragma unroll
            for (int b = 0; b < NB; ++b) {        // softmax over b (axis=-2)
                v[b] = row[b * NC + c];
                mx   = fmaxf(mx, v[b]);
            }
            float s = 0.0f;
            #pragma unroll
            for (int b = 0; b < NB; ++b) {
                v[b] = __expf(v[b] - mx);
                s += v[b];
            }
            const float inv = __builtin_amdgcn_rcpf(s);
            #pragma unroll
            for (int b = 0; b < NB; ++b) {
                w_lds[mloc][b * NC + c] = (_Float16)(v[b] * inv);
            }
        }
    }
    __syncthreads();   // the only barrier in the kernel

    // ---------------- Phase 2: WMMA GEMM, barrier-free inner loop ---------------
    const int wave    = t >> 5;              // 0..7 ; wave owns N cols [wave*32, wave*32+32)
    const int lane    = t & 31;
    const int nhalf   = lane & 15;
    const int n0      = wave * 32;
    const int khalf_a = (lane >> 4) * 8;     // A: elems 0..7 -> k+khalf, 8..15 -> k+16+khalf
    const int koff_b  = (lane >> 4) * 8;     // B: elems 0..7 -> k+koff_b, 8..15 -> k+16+koff_b

    const _Float16* arow0 = &w_lds[(lane & 15)     ][khalf_a];   // M-tile 0
    const _Float16* arow1 = &w_lds[(lane & 15) + 16][khalf_a];   // M-tile 1
    const _Float16* brow0 = cbT + (size_t)(n0 + nhalf)      * KDIM + koff_b;
    const _Float16* brow1 = cbT + (size_t)(n0 + 16 + nhalf) * KDIM + koff_b;

    v8f acc00 = {}, acc01 = {}, acc10 = {}, acc11 = {};

    #pragma unroll 2
    for (int kt = 0; kt < KDIM / 32; ++kt) {
        const int k0 = kt * 32;

        // A fragments (16x32 f16) from LDS: conflict-free ds_load_b128 pairs.
        const v8h a0lo = *(const v8h*)(arow0 + k0);
        const v8h a0hi = *(const v8h*)(arow0 + k0 + 16);
        const v16h a0  = __builtin_shufflevector(a0lo, a0hi,
            0,1,2,3,4,5,6,7,8,9,10,11,12,13,14,15);
        const v8h a1lo = *(const v8h*)(arow1 + k0);
        const v8h a1hi = *(const v8h*)(arow1 + k0 + 16);
        const v16h a1  = __builtin_shufflevector(a1lo, a1hi,
            0,1,2,3,4,5,6,7,8,9,10,11,12,13,14,15);

        // B fragments (32x16 f16): two 8-f16 runs (k, k+16) straight from L2-resident cbT.
        const v8h b0lo = *(const v8h*)(brow0 + k0);
        const v8h b0hi = *(const v8h*)(brow0 + k0 + 16);
        const v16h b0  = __builtin_shufflevector(b0lo, b0hi,
            0,1,2,3,4,5,6,7,8,9,10,11,12,13,14,15);
        const v8h b1lo = *(const v8h*)(brow1 + k0);
        const v8h b1hi = *(const v8h*)(brow1 + k0 + 16);
        const v16h b1  = __builtin_shufflevector(b1lo, b1hi,
            0,1,2,3,4,5,6,7,8,9,10,11,12,13,14,15);

        acc00 = __builtin_amdgcn_wmma_f32_16x16x32_f16(
            false, a0, false, b0, (short)0, acc00, false, false);
        acc01 = __builtin_amdgcn_wmma_f32_16x16x32_f16(
            false, a0, false, b1, (short)0, acc01, false, false);
        acc10 = __builtin_amdgcn_wmma_f32_16x16x32_f16(
            false, a1, false, b0, (short)0, acc10, false, false);
        acc11 = __builtin_amdgcn_wmma_f32_16x16x32_f16(
            false, a1, false, b1, (short)0, acc11, false, false);
    }

    // ---------------- Store: D layout -> global (coalesced across lanes 0-15) ---
    // VGPR r: lanes 0-15 hold M=r, N=lane; lanes 16-31 hold M=8+r, N=lane-16.
    const int mrow = (lane >> 4) * 8;
    float* __restrict__ outp = out + (size_t)blk * MTILE * DDIM;
    #pragma unroll
    for (int r = 0; r < 8; ++r) {
        const int m0 = mrow + r;          // M-tile 0 rows
        const int m1 = 16 + mrow + r;     // M-tile 1 rows
        outp[(size_t)m0 * DDIM + n0 + nhalf]      = acc00[r];
        outp[(size_t)m0 * DDIM + n0 + 16 + nhalf] = acc01[r];
        outp[(size_t)m1 * DDIM + n0 + nhalf]      = acc10[r];
        outp[(size_t)m1 * DDIM + n0 + 16 + nhalf] = acc11[r];
    }
}

extern "C" void kernel_launch(void* const* d_in, const int* in_sizes, int n_in,
                              void* d_out, int out_size, void* d_ws, size_t ws_size,
                              hipStream_t stream) {
    const int*   tokens   = (const int*)d_in[0];    // (64, 512) int32
    const float* code     = (const float*)d_in[1];  // (50000, 32, 64) f32
    const float* codebook = (const float*)d_in[2];  // (32, 64, 256) f32 = (2048, 256)
    float*       out      = (float*)d_out;          // (64, 512, 256) f32

    _Float16* cbT = (_Float16*)d_ws;                // 256 x 2048 f16 = 1 MB scratch

    // 1) One-time (per launch) convert+transpose of the codebook into scratch.
    convert_codebook_kernel<<<dim3((DDIM * KDIM / 8) / 256), dim3(256), 0, stream>>>(
        codebook, cbT);

    // 2) Fused gather/softmax/GEMM. Same stream -> ordered after the convert.
    const int M = in_sizes[0];                      // 32768 tokens
    comp_embed_fused_kernel<<<dim3(M / MTILE), dim3(256), 0, stream>>>(
        tokens, code, cbT, out);

    (void)ws_size; (void)n_in; (void)out_size;
}
